// NearestEmbed_20529943675246
// MI455X (gfx1250) — matrix operand order repaired and checked by reading, not compiled, not run on hardware
//
#include <hip/hip_runtime.h>

// ---- CDNA5 WMMA types -------------------------------------------------------
typedef __attribute__((ext_vector_type(16))) __bf16 bf16x16;
typedef __attribute__((ext_vector_type(8)))  float  f32x8;

// Problem constants: x (32,256,32,32) f32, emb (256,2048) f32
#define NB       32
#define ND       256
#define NHW      1024
#define NN       (NB*NHW)      // 32768 latents
#define NK       2048          // codes
#define KT_N     (NK/16)       // 128 column tiles
#define DT_N     (ND/32)       // 8 K-chunks of 32 per WMMA
#define BLK_ROWS 64            // latent rows per block (4 waves x 16)
#define NTHR     128           // 4 waves

// Workspace layout (bytes)
#define WS_EMBN_OFF   0                        // 2048 f32 = 8 KB
#define WS_FRAG_OFF   8192                     // swizzled emb frags, 2 MB
#define WS_FRAG_BYTES (KT_N*DT_N*2*32*16*2)
#define WS_ARG_OFF    (WS_FRAG_OFF + WS_FRAG_BYTES)   // 32768 int

// ---- CDNA5 async global->LDS path (probe via __has_builtin) -----------------
#if defined(__has_builtin)
# if __has_builtin(__builtin_amdgcn_global_load_async_to_lds_b128)
#  define HAVE_ASYNC_LDS 1
# endif
#endif

// Builtin parameter pointee type per hipcc diagnostic:
//   'int __attribute__((__vector_size__(4 * sizeof(int)))) *'
typedef int v4i __attribute__((vector_size(16)));
typedef __attribute__((address_space(1))) v4i v4i_g;   // global
typedef __attribute__((address_space(3))) v4i v4i_l;   // LDS

template <int N>
__device__ __forceinline__ void waitAsyncN() {
#if defined(HAVE_ASYNC_LDS)
# if __has_builtin(__builtin_amdgcn_s_wait_asynccnt)
    __builtin_amdgcn_s_wait_asynccnt((unsigned short)N);
# else
    if (N == 0) asm volatile("s_wait_asynccnt 0x0" ::: "memory");
    else        asm volatile("s_wait_asynccnt 0x8" ::: "memory");
# endif
#endif
}

// Copy one 16KB B k-tile (1024 uint4) global -> LDS. Async if available.
__device__ __forceinline__ void issueB(uint4* __restrict__ dst,
                                       const uint4* __restrict__ src, int tid) {
#if defined(HAVE_ASYNC_LDS)
#pragma unroll
    for (int i = 0; i < 8; ++i) {
        int j = tid + i * NTHR;
        __builtin_amdgcn_global_load_async_to_lds_b128(
            (v4i_g*)(src + j), (v4i_l*)(dst + j), 0, 0);
    }
#else
    for (int i = tid; i < 1024; i += NTHR) dst[i] = src[i];
#endif
}

// ---- Kernel 1: per-code squared norms --------------------------------------
__global__ void vq_embn_kernel(const float* __restrict__ emb, float* __restrict__ embn) {
    int k = blockIdx.x * 256 + threadIdx.x;
    float acc = 0.f;
    for (int d = 0; d < ND; ++d) {
        float v = emb[(size_t)d * NK + k];
        acc += v * v;
    }
    embn[k] = acc;
}

// ---- Kernel 2: pre-swizzle emb into per-lane WMMA B fragments ---------------
// frag[kt][dt][which(hi,lo)][lane][e]; B (32x16 bf16): lanes 0-15 hold d=0..15,
// lanes 16-31 hold d=16..31; e = kk & 15, lane = n + 16*(kk>>4).
__global__ void vq_fragB_kernel(const float* __restrict__ emb, __bf16* __restrict__ frag) {
    unsigned idx = blockIdx.x * 256 + threadIdx.x;
    int e     =  idx        & 15;
    int lane  = (idx >> 4)  & 31;
    int which = (idx >> 9)  & 1;
    int dt    = (idx >> 10) & 7;
    int kt    =  idx >> 13;
    int n  = lane & 15;
    int kk = e + 16 * (lane >> 4);
    float v = emb[(size_t)(dt * 32 + kk) * NK + (kt * 16 + n)];
    __bf16 h = (__bf16)v;
    frag[idx] = which ? (__bf16)(v - (float)h) : h;
}

// ---- Kernel 3: WMMA distance + running argmin -------------------------------
__global__ __launch_bounds__(NTHR) void vq_main_kernel(
    const float* __restrict__ x,
    const __bf16* __restrict__ embFrag,
    const float* __restrict__ embn,
    int* __restrict__ argminOut)
{
    __shared__ __bf16 ldsB[2][DT_N * 2 * 512];  // double-buffered B: 2 x 16 KB
    __shared__ __bf16 ldsStage[4 * 2 * 512];    // A transpose scratch: 8 KB

    const int tid  = threadIdx.x;
    const int lane = tid & 31;
    const int wave = tid >> 5;
    const int nBase = blockIdx.x * BLK_ROWS;
    const uint4* fragSrc = (const uint4*)embFrag;   // 1024 uint4 per k-tile

    // Prefetch first B tile immediately; it overlaps the whole A staging phase.
    issueB((uint4*)ldsB[0], fragSrc, tid);

    // ---- Build per-wave A fragments in REGISTERS (hi+lo split bf16) ----
    // A (16x32 bf16): half=kk>>4, lane=m+16*half, e=kk&15 (matches B fragment k order).
    bf16x16 Ahi[DT_N], Alo[DT_N];
#pragma unroll
    for (int dt = 0; dt < DT_N; ++dt) {
        for (int i = tid; i < BLK_ROWS * 32; i += NTHR) {
            int nl = i & (BLK_ROWS - 1);
            int kk = i >> 6;
            int n  = nBase + nl;
            int b  = n >> 10;
            int hw = n & (NHW - 1);
            float v = x[(size_t)b * (ND * NHW) + (size_t)(dt * 32 + kk) * NHW + hw];
            __bf16 h = (__bf16)v;
            __bf16 l = (__bf16)(v - (float)h);
            int w    = nl >> 4;
            int m    = nl & 15;
            int half = kk >> 4;
            int ln   = m + 16 * half;
            int e    = kk & 15;
            ldsStage[(w * 2 + 0) * 512 + ln * 16 + e] = h;
            ldsStage[(w * 2 + 1) * 512 + ln * 16 + e] = l;
        }
        __syncthreads();
        Ahi[dt] = *(const bf16x16*)&ldsStage[(wave * 2 + 0) * 512 + lane * 16];
        Alo[dt] = *(const bf16x16*)&ldsStage[(wave * 2 + 1) * 512 + lane * 16];
        __syncthreads();
    }

    float best[8];
    int   bestk[8];
#pragma unroll
    for (int v = 0; v < 8; ++v) { best[v] = 3.4e38f; bestk[v] = 0; }

    for (int kt = 0; kt < KT_N; ++kt) {
        const int cur = kt & 1;
        if (kt + 1 < KT_N) {
            issueB((uint4*)ldsB[cur ^ 1], fragSrc + (size_t)(kt + 1) * 1024, tid);
            waitAsyncN<8>();   // 8 newest in flight = next buffer; current 8 are done
        } else {
            waitAsyncN<0>();
        }
        __syncthreads();

        float en   = embn[kt * 16 + (lane & 15)];
        int   kcol = kt * 16 + (lane & 15);

        // 16x16 dot tile: 8 d-chunks x 3 split-bf16 WMMAs, 2 accumulators for ILP
        f32x8 acc0 = {}, acc1 = {};
        const __bf16* Bb = ldsB[cur];
#pragma unroll
        for (int dt = 0; dt < DT_N; ++dt) {
            const bf16x16 b_hi = *(const bf16x16*)&Bb[(dt * 2 + 0) * 512 + lane * 16];
            const bf16x16 b_lo = *(const bf16x16*)&Bb[(dt * 2 + 1) * 512 + lane * 16];
            if (dt & 1) {
                acc1 = __builtin_amdgcn_wmma_f32_16x16x32_bf16(false, Ahi[dt], false, b_hi, (short)0, acc1, false, false);
                acc1 = __builtin_amdgcn_wmma_f32_16x16x32_bf16(false, Ahi[dt], false, b_lo, (short)0, acc1, false, false);
                acc1 = __builtin_amdgcn_wmma_f32_16x16x32_bf16(false, Alo[dt], false, b_hi, (short)0, acc1, false, false);
            } else {
                acc0 = __builtin_amdgcn_wmma_f32_16x16x32_bf16(false, Ahi[dt], false, b_hi, (short)0, acc0, false, false);
                acc0 = __builtin_amdgcn_wmma_f32_16x16x32_bf16(false, Ahi[dt], false, b_lo, (short)0, acc0, false, false);
                acc0 = __builtin_amdgcn_wmma_f32_16x16x32_bf16(false, Alo[dt], false, b_hi, (short)0, acc0, false, false);
            }
        }

        // score = ||e_k||^2 - 2*dot ; C/D layout: lane = N column, VGPR v = M row
#pragma unroll
        for (int v = 0; v < 8; ++v) {
            float s = en - 2.0f * (acc0[v] + acc1[v]);
            if (s < best[v]) { best[v] = s; bestk[v] = kcol; }
        }
        __syncthreads();   // all waves done reading cur before it is refilled
    }

    // ---- reduce min across the 16 columns held by each 16-lane half ----
#pragma unroll
    for (int off = 1; off <= 8; off <<= 1) {
#pragma unroll
        for (int v = 0; v < 8; ++v) {
            float ov = __shfl_xor(best[v], off, 32);
            int   oi = __shfl_xor(bestk[v], off, 32);
            if (ov < best[v] || (ov == best[v] && oi < bestk[v])) {
                best[v] = ov; bestk[v] = oi;
            }
        }
    }
    if ((lane & 15) == 0) {
        int rowBase = nBase + wave * 16 + ((lane >> 4) ? 8 : 0);
#pragma unroll
        for (int v = 0; v < 8; ++v) argminOut[rowBase + v] = bestk[v];
    }
}

// ---- Kernel 4: gather winning codes into (B, D, H, W) -----------------------
__global__ void vq_gather_kernel(const float* __restrict__ emb,
                                 const int* __restrict__ argmin,
                                 float* __restrict__ out)
{
    unsigned idx = blockIdx.x * 256 + threadIdx.x;
    int n = idx & (NN - 1);
    int d = idx >> 15;
    int b = n >> 10, hw = n & (NHW - 1);
    out[(size_t)b * (ND * NHW) + (size_t)d * NHW + hw] = emb[(size_t)d * NK + argmin[n]];
}

// ---- Launcher ---------------------------------------------------------------
extern "C" void kernel_launch(void* const* d_in, const int* in_sizes, int n_in,
                              void* d_out, int out_size, void* d_ws, size_t ws_size,
                              hipStream_t stream) {
    const float* x   = (const float*)d_in[0];
    const float* emb = (const float*)d_in[1];
    float* out = (float*)d_out;

    char* ws = (char*)d_ws;
    float*  embn = (float*)(ws + WS_EMBN_OFF);
    __bf16* frag = (__bf16*)(ws + WS_FRAG_OFF);
    int*    amin = (int*)(ws + WS_ARG_OFF);

    vq_embn_kernel  <<<NK / 256, 256, 0, stream>>>(emb, embn);
    vq_fragB_kernel <<<(KT_N * DT_N * 2 * 32 * 16) / 256, 256, 0, stream>>>(emb, frag);
    vq_main_kernel  <<<NN / BLK_ROWS, NTHR, 0, stream>>>(x, frag, embn, amin);
    vq_gather_kernel<<<(NN * ND) / 256, 256, 0, stream>>>(emb, amin, out);
}
